// Gemma3Attention_16355235463291
// MI455X (gfx1250) — compile-verified, matrix operand order checked by reading, not run on hardware
//
#include <hip/hip_runtime.h>
#include <hip/hip_fp16.h>

typedef __attribute__((ext_vector_type(16))) _Float16 v16h;
typedef __attribute__((ext_vector_type(8)))  float    v8f;

constexpr int HDIM  = 256;
constexpr int NH    = 8;
constexpr int NKV   = 4;
constexpr int SEQ   = 2048;
constexpr int BATCH = 2;
constexpr int HID   = 2560;
constexpr int QKV_N = NH * HDIM + 2 * NKV * HDIM;   // 4096
constexpr int Q_SIZE = NH * HDIM;                   // 2048
constexpr float SCALEF = 0.0625f;                   // 256^-0.5
constexpr int WINDOW = 1023;

// ---------------------------------------------------------------------------
// WMMA fragment loader per CDNA5 ISA 7.12.2 (wave32, 16-bit operands).
// A 16x32: lane m = L&15, hi = L>>4; VGPR p holds the k-pair {k, k+1} with
//          k = (p&4)*4 + (p&3)*2 + hi*8  -> packed dword loads (merge to b128)
// With B tiles staged TRANSPOSED in LDS (Bt[n][k]), B fragments use the same
// packed loader (row selector = n), so no scalar strided LDS reads anywhere.
// C/D 16x16: lane n = L&15; VGPR r -> m = r + hi*8
// ---------------------------------------------------------------------------
static __device__ __forceinline__ v16h load_frag16(const _Float16* base, int pitch, int lane) {
  const int m  = lane & 15;
  const int hi = lane >> 4;
  v16h a;
  unsigned int* ap = reinterpret_cast<unsigned int*>(&a);
  const _Float16* row = base + m * pitch + (hi << 3);
#pragma unroll
  for (int p = 0; p < 8; ++p) {
    const int k = ((p & 4) << 2) + ((p & 3) << 1);
    ap[p] = *reinterpret_cast<const unsigned int*>(row + k);
  }
  return a;
}

static __device__ __forceinline__ v8f wmma_f16(v16h a, v16h b, v8f c) {
  // D = A(16x32 f16) * B(32x16 f16) + C(16x16 f32)
  return __builtin_amdgcn_wmma_f32_16x16x32_f16(false, a, false, b, (short)0, c, false, false);
}

// Async global->LDS 16-byte copy (GLOBAL_LOAD_ASYNC_TO_LDS_B128, ASYNCcnt).
static __device__ __forceinline__ void async_g2l_b128(unsigned lds_byte_off, const void* gptr) {
  asm volatile("global_load_async_to_lds_b128 %0, %1, off"
               :: "v"(lds_byte_off), "v"(gptr)
               : "memory");
}
static __device__ __forceinline__ void wait_async0() {
  asm volatile("s_wait_asynccnt 0x0" ::: "memory");
}

// ---------------------------------------------------------------------------
// Generic f32-in / f32-out GEMM via f16 WMMA. C[M,N] = A[M,K] @ B[K,N].
// Block: 256 thr (8 waves 2x4); tile 128x128x32; wave tile 64x32.
// Double-buffered LDS; B tile staged transposed (Bt[n][k]).
// ---------------------------------------------------------------------------
constexpr int G_BM = 128, G_BN = 128, G_BK = 32;
constexpr int G_AP = G_BK + 8;                      // pitch 40

static __device__ __forceinline__ void gemm_stage(
    _Float16* __restrict__ As, _Float16* __restrict__ Bt,
    const float* __restrict__ A, const float* __restrict__ Bm,
    int M, int N, int K, int m0, int n0, int k0, int tid) {
  // A tile 128x32: float4 per thread, pack 4 cvt halves -> one b64 store
#pragma unroll
  for (int i = tid * 4; i < G_BM * G_BK; i += 256 * 4) {
    const int r = i >> 5, c = i & (G_BK - 1);
    float4 v = make_float4(0.f, 0.f, 0.f, 0.f);
    if (m0 + r < M)
      v = *reinterpret_cast<const float4*>(A + (size_t)(m0 + r) * K + k0 + c);
    union { _Float16 h[4]; uint2 u; } pk;
    pk.h[0] = (_Float16)v.x; pk.h[1] = (_Float16)v.y;
    pk.h[2] = (_Float16)v.z; pk.h[3] = (_Float16)v.w;
    *reinterpret_cast<uint2*>(&As[r * G_AP + c]) = pk.u;
  }
  // B tile 32x128 staged transposed: thread owns column n, half of k range.
  const int n   = tid & 127;
  const int kq0 = (tid >> 7) * 16;    // 0 or 16
  const bool ok = (n0 + n) < N;
#pragma unroll
  for (int kq = 0; kq < 16; kq += 4) {
    union { _Float16 h[4]; uint2 u; } pk;
#pragma unroll
    for (int j = 0; j < 4; ++j) {
      const int k = kq0 + kq + j;
      pk.h[j] = ok ? (_Float16)Bm[(size_t)(k0 + k) * N + n0 + n] : (_Float16)0.f;
    }
    *reinterpret_cast<uint2*>(&Bt[n * G_AP + kq0 + kq]) = pk.u;
  }
}

__global__ __launch_bounds__(256) void gemm_wmma_f16(
    const float* __restrict__ A, const float* __restrict__ Bm, float* __restrict__ C,
    int M, int N, int K) {
  __shared__ __align__(16) _Float16 As[2][G_BM * G_AP];
  __shared__ __align__(16) _Float16 Bt[2][G_BN * G_AP];
  const int tid = threadIdx.x, lane = tid & 31, wave = tid >> 5;
  const int wm = wave >> 2, wn = wave & 3;
  const int m0 = blockIdx.y * G_BM, n0 = blockIdx.x * G_BN;
  v8f acc[4][2] = {};

  const int KT = K / G_BK;
  gemm_stage(As[0], Bt[0], A, Bm, M, N, K, m0, n0, 0, tid);
  __syncthreads();
  for (int kt = 0; kt < KT; ++kt) {
    const int cur = kt & 1;
    if (kt + 1 < KT)
      gemm_stage(As[1 - cur], Bt[1 - cur], A, Bm, M, N, K, m0, n0, (kt + 1) * G_BK, tid);
    const _Float16* Asb = &As[cur][0];
    const _Float16* Btb = &Bt[cur][0];
    v16h bf[2];
#pragma unroll
    for (int nf = 0; nf < 2; ++nf)
      bf[nf] = load_frag16(Btb + (wn * 32 + nf * 16) * G_AP, G_AP, lane);
#pragma unroll
    for (int mf = 0; mf < 4; ++mf) {
      const v16h af = load_frag16(Asb + (wm * 64 + mf * 16) * G_AP, G_AP, lane);
#pragma unroll
      for (int nf = 0; nf < 2; ++nf)
        acc[mf][nf] = wmma_f16(af, bf[nf], acc[mf][nf]);
    }
    __syncthreads();
  }

  const int hi = lane >> 4, nn = lane & 15;
#pragma unroll
  for (int mf = 0; mf < 4; ++mf)
#pragma unroll
    for (int nf = 0; nf < 2; ++nf)
#pragma unroll
      for (int r = 0; r < 8; ++r) {
        const int row = m0 + wm * 64 + mf * 16 + r + (hi << 3);
        const int col = n0 + wn * 32 + nf * 16 + nn;
        if (row < M && col < N)
          C[(size_t)row * N + col] = acc[mf][nf][r];
      }
}

// ---------------------------------------------------------------------------
// Per-token RMSNorm (q/k heads) + RoPE, V passthrough; f32 qkv -> f16 Q/K/V
// in [B, H, S, HD] layout. 1 block / token, 8 waves x 2 head-rows.
// ---------------------------------------------------------------------------
__global__ __launch_bounds__(256) void normrope_kernel(
    const float* __restrict__ qkv, const int* __restrict__ positions,
    const float* __restrict__ qw, const float* __restrict__ kw,
    _Float16* __restrict__ Qh, _Float16* __restrict__ Kh, _Float16* __restrict__ Vh) {
  const int t = blockIdx.x;
  const int b = t / SEQ, s = t - b * SEQ;
  const int lane = threadIdx.x & 31, wave = threadIdx.x >> 5;
  const float* row = qkv + (size_t)t * QKV_N;
  const float pos = (float)positions[t];
  for (int rr = 0; rr < 2; ++rr) {
    const int ri = wave * 2 + rr;
    const float* src;
    _Float16* dst;
    const float* w = nullptr;
    bool dorope = true;
    if (ri < NH) {
      src = row + ri * HDIM;
      dst = Qh + ((size_t)(b * NH + ri) * SEQ + s) * HDIM;
      w = qw;
    } else if (ri < NH + NKV) {
      const int hh = ri - NH;
      src = row + Q_SIZE + hh * HDIM;
      dst = Kh + ((size_t)(b * NKV + hh) * SEQ + s) * HDIM;
      w = kw;
    } else {
      const int hh = ri - NH - NKV;
      src = row + Q_SIZE + NKV * HDIM + hh * HDIM;
      dst = Vh + ((size_t)(b * NKV + hh) * SEQ + s) * HDIM;
      dorope = false;
    }
    if (!dorope) {
#pragma unroll
      for (int i = 0; i < 8; ++i) {
        const int idx = lane * 8 + i;
        dst[idx] = (_Float16)src[idx];
      }
      continue;
    }
    float x1[4], x2[4];
    float ss = 0.f;
#pragma unroll
    for (int i = 0; i < 4; ++i) {
      const int j = lane * 4 + i;
      x1[i] = src[j];
      x2[i] = src[j + 128];
      ss += x1[i] * x1[i] + x2[i] * x2[i];
    }
    ss += __shfl_xor(ss, 1);
    ss += __shfl_xor(ss, 2);
    ss += __shfl_xor(ss, 4);
    ss += __shfl_xor(ss, 8);
    ss += __shfl_xor(ss, 16);
    const float inv = rsqrtf(ss * (1.0f / 256.0f) + 1e-6f);
#pragma unroll
    for (int i = 0; i < 4; ++i) {
      const int j = lane * 4 + i;
      const float n1 = x1[i] * inv * (1.0f + w[j]);
      const float n2 = x2[i] * inv * (1.0f + w[j + 128]);
      // inv_freq = 10000^(-2j/256)
      const float fr = pos * __expf(-(float)(2 * j) * (9.2103403719761836f / 256.0f));
      const float c = __cosf(fr), sn = __sinf(fr);
      dst[j]       = (_Float16)(n1 * c - n2 * sn);
      dst[j + 128] = (_Float16)(n2 * c + n1 * sn);
    }
  }
}

// ---------------------------------------------------------------------------
// Sliding-window flash attention, GQA (2 q heads / kv head), WMMA f16.
// Block: 128 thr (4 waves x 16 queries).
//   K block (64x256, row-major)  : staged via GLOBAL_LOAD_ASYNC_TO_LDS_B128
//   V block (staged TRANSPOSED)  : Vt[d][key] -> packed b-frag loads for PV
// K and Vt share one flat LDS buffer (union in time); P spilled per wave.
// Output: attn[B, S, NH, HD] f32.
// ---------------------------------------------------------------------------
constexpr int A_BQ = 64, A_BKV = 64;
constexpr int A_KP = HDIM + 8;      // 264: K rows (key-major, d contiguous)
constexpr int A_VP = A_BKV + 8;     // 72:  Vt rows (d-major, key contiguous)
constexpr int A_KV_ELEMS = (A_BKV * A_KP) > (HDIM * A_VP) ? (A_BKV * A_KP) : (HDIM * A_VP);

__global__ __launch_bounds__(128) void attn_kernel(
    const _Float16* __restrict__ Qh, const _Float16* __restrict__ Kh,
    const _Float16* __restrict__ Vh, const int* __restrict__ amask,
    float* __restrict__ attn) {
  __shared__ __align__(16) _Float16 kv[A_KV_ELEMS];                 // 36 KB
  __shared__ __align__(16) _Float16 Ps[4][16 * A_VP];               // 9 KB
  __shared__ float Lmask[A_BKV];
  const int tid = threadIdx.x, lane = tid & 31, wave = tid >> 5;
  const int qb = blockIdx.x, h = blockIdx.y, b = blockIdx.z;
  const int kvh = h >> 1;                       // g = NH/NKV = 2
  const int q0 = qb * A_BQ;
  const int qrow = q0 + wave * 16;
  const _Float16* Qbase = Qh + ((size_t)(b * NH + h) * SEQ + qrow) * HDIM;
  const _Float16* Kbase = Kh + (size_t)(b * NKV + kvh) * SEQ * HDIM;
  const _Float16* Vbase = Vh + (size_t)(b * NKV + kvh) * SEQ * HDIM;
  const int hi = lane >> 4, nn = lane & 15;
  const unsigned kvbase = (unsigned)(uintptr_t)(&kv[0]);

  v16h qf[8];
#pragma unroll
  for (int kf = 0; kf < 8; ++kf)
    qf[kf] = load_frag16(Qbase + kf * 32, HDIM, lane);

  v8f O[16] = {};
  float mrun[8], lrun[8];
#pragma unroll
  for (int r = 0; r < 8; ++r) { mrun[r] = -1e30f; lrun[r] = 0.f; }

  int klo = q0 - WINDOW;
  if (klo < 0) klo = 0;
  klo &= ~(A_BKV - 1);
  const int khi = q0 + A_BQ;

  for (int kb = klo; kb < khi; kb += A_BKV) {
    __syncthreads();
    // --- stage K block async (raw f16 copy, bypasses VGPRs; ASYNCcnt) ---
    for (int i = tid * 8; i < A_BKV * HDIM; i += 128 * 8) {
      const int r = i >> 8, c = i & (HDIM - 1);
      const int key = kb + r;
      if (key < SEQ) {
        async_g2l_b128(kvbase + (unsigned)((r * A_KP + c) * 2),
                       Kbase + (size_t)key * HDIM + c);
      } else {
        *reinterpret_cast<uint4*>(&kv[r * A_KP + c]) = make_uint4(0u, 0u, 0u, 0u);
      }
    }
    if (tid < A_BKV) {
      const int key = kb + tid;
      Lmask[tid] = (key < SEQ && amask[b * SEQ + key] != 0) ? 0.0f : -1e30f;
    }
    wait_async0();
    __syncthreads();

    // --- scores S = Q (16x256) @ K^T (256x64): 4 n-frags x 8 k-steps ---
    v8f Sf[4];
#pragma unroll
    for (int nf = 0; nf < 4; ++nf) {
      v8f s = {};
#pragma unroll
      for (int kf = 0; kf < 8; ++kf) {
        // B[k][n] = K[key = nf*16+n][d = kf*32+k]: d contiguous -> packed frag
        const v16h bk = load_frag16(&kv[(nf * 16) * A_KP + kf * 32], A_KP, lane);
        s = wmma_f16(qf[kf], bk, s);
      }
      Sf[nf] = s;
    }

    // --- scale + causal/window/pad mask (C-frag coords) ---
#pragma unroll
    for (int nf = 0; nf < 4; ++nf) {
      const int key = kb + nf * 16 + nn;
      const float madd = Lmask[nf * 16 + nn];
#pragma unroll
      for (int r = 0; r < 8; ++r) {
        const int q = qrow + r + (hi << 3);
        float sv = Sf[nf][r] * SCALEF + madd;
        if (key > q || (q - key) > WINDOW) sv = -1e30f;
        Sf[nf][r] = sv;
      }
    }

    // --- online softmax: row lives in one 16-lane half -> xor {1,2,4,8} ---
#pragma unroll
    for (int r = 0; r < 8; ++r) {
      float mx = fmaxf(fmaxf(Sf[0][r], Sf[1][r]), fmaxf(Sf[2][r], Sf[3][r]));
      mx = fmaxf(mx, __shfl_xor(mx, 1));
      mx = fmaxf(mx, __shfl_xor(mx, 2));
      mx = fmaxf(mx, __shfl_xor(mx, 4));
      mx = fmaxf(mx, __shfl_xor(mx, 8));
      const float newm = fmaxf(mrun[r], mx);
      const float corr = __expf(mrun[r] - newm);
      mrun[r] = newm;
      float ps = 0.f;
#pragma unroll
      for (int nf = 0; nf < 4; ++nf) {
        const float sv = Sf[nf][r];
        float p = __expf(sv - newm);
        p = (sv > -1e29f) ? p : 0.0f;   // fully-masked-row guard
        Sf[nf][r] = p;
        ps += p;
      }
      ps += __shfl_xor(ps, 1);
      ps += __shfl_xor(ps, 2);
      ps += __shfl_xor(ps, 4);
      ps += __shfl_xor(ps, 8);
      lrun[r] = lrun[r] * corr + ps;
#pragma unroll
      for (int df = 0; df < 16; ++df) O[df][r] *= corr;
    }

    // --- spill P (C-frag layout) to per-wave LDS for A-frag reload ---
#pragma unroll
    for (int nf = 0; nf < 4; ++nf)
#pragma unroll
      for (int r = 0; r < 8; ++r)
        Ps[wave][(r + (hi << 3)) * A_VP + nf * 16 + nn] = (_Float16)Sf[nf][r];

    __syncthreads();
    // --- stage V block TRANSPOSED into the same buffer: Vt[d][key] ---
    // thread owns d rows; 4 consecutive keys pack into one b64 store.
    for (int d = tid; d < HDIM; d += 128) {
#pragma unroll
      for (int k4 = 0; k4 < A_BKV; k4 += 4) {
        union { _Float16 h[4]; uint2 u; } pk;
#pragma unroll
        for (int j = 0; j < 4; ++j) {
          const int key = kb + k4 + j;
          pk.h[j] = (key < SEQ) ? Vbase[(size_t)key * HDIM + d] : (_Float16)0.f;
        }
        *reinterpret_cast<uint2*>(&kv[d * A_VP + k4]) = pk.u;
      }
    }
    __syncthreads();

    // --- O += P (16x64) @ V (64x256): 2 k-steps x 16 d-frags, all packed ---
#pragma unroll
    for (int ks = 0; ks < 2; ++ks) {
      const v16h pf = load_frag16(&Ps[wave][ks * 32], A_VP, lane);
#pragma unroll
      for (int df = 0; df < 16; ++df) {
        // B[k][n] = V[key = ks*32+k][d = df*16+n] = Vt[d][key]: key contiguous
        const v16h vf = load_frag16(&kv[(df * 16) * A_VP + ks * 32], A_VP, lane);
        O[df] = wmma_f16(pf, vf, O[df]);
      }
    }
  }

  // --- normalize + store f32 to attn[B, S, NH, HD] ---
  float* outp = attn + (((size_t)b * SEQ + qrow) * NH + h) * HDIM;
#pragma unroll
  for (int r = 0; r < 8; ++r) {
    const float rl = (lrun[r] > 0.f) ? (1.0f / lrun[r]) : 0.0f;
    const int qr = r + (hi << 3);
#pragma unroll
    for (int df = 0; df < 16; ++df)
      outp[(size_t)qr * (NH * HDIM) + df * 16 + nn] = O[df][r] * rl;
  }
}

// ---------------------------------------------------------------------------
extern "C" void kernel_launch(void* const* d_in, const int* in_sizes, int n_in,
                              void* d_out, int out_size, void* d_ws, size_t ws_size,
                              hipStream_t stream) {
  (void)in_sizes; (void)n_in; (void)out_size; (void)ws_size;
  const int*   positions = (const int*)d_in[0];
  const float* hidden    = (const float*)d_in[1];
  const int*   amask     = (const int*)d_in[2];
  const float* Wqkv      = (const float*)d_in[3];
  const float* Wo        = (const float*)d_in[4];
  const float* qw        = (const float*)d_in[5];
  const float* kw        = (const float*)d_in[6];
  float* out = (float*)d_out;

  const int BS = BATCH * SEQ;                       // 4096
  char* ws = (char*)d_ws;
  size_t off = 0;
  float*    qkv  = (float*)(ws + off);    off += (size_t)BS * QKV_N * sizeof(float);
  _Float16* Qh   = (_Float16*)(ws + off); off += (size_t)BS * Q_SIZE * sizeof(_Float16);
  _Float16* Kh   = (_Float16*)(ws + off); off += (size_t)BS * NKV * HDIM * sizeof(_Float16);
  _Float16* Vh   = (_Float16*)(ws + off); off += (size_t)BS * NKV * HDIM * sizeof(_Float16);
  float*    attn = (float*)(ws + off);    off += (size_t)BS * Q_SIZE * sizeof(float);

  // 1) QKV projection: [4096,2560] @ [2560,4096]
  gemm_wmma_f16<<<dim3(QKV_N / 128, BS / 128), 256, 0, stream>>>(
      hidden, Wqkv, qkv, BS, QKV_N, HID);

  // 2) RMSNorm + RoPE -> f16 Q/K/V in [B,H,S,HD]
  normrope_kernel<<<BS, 256, 0, stream>>>(qkv, positions, qw, kw, Qh, Kh, Vh);

  // 3) Sliding-window flash attention
  attn_kernel<<<dim3(SEQ / 64, NH, BATCH), 128, 0, stream>>>(Qh, Kh, Vh, amask, attn);

  // 4) Output projection: [4096,2048] @ [2048,2560]
  gemm_wmma_f16<<<dim3(HID / 128, BS / 128), 256, 0, stream>>>(
      attn, Wo, out, BS, HID, Q_SIZE);
}
